// KANLayer_50611894616756
// MI455X (gfx1250) — compile-verified, matrix operand order checked by reading, not run on hardware
//
#include <hip/hip_runtime.h>
#include <hip/hip_bf16.h>
#include <stdint.h>

typedef __attribute__((ext_vector_type(16))) _Float16 v16h;
typedef __attribute__((ext_vector_type(8)))  _Float16 v8h;
typedef __attribute__((ext_vector_type(8)))  float    v8f;
typedef __attribute__((ext_vector_type(4)))  float    f32x4;
typedef __attribute__((ext_vector_type(4)))  unsigned u32x4;
typedef __attribute__((ext_vector_type(8)))  unsigned u32x8;

namespace {
constexpr int IN_DIM   = 256;
constexpr int OUT_DIM  = 256;
constexpr int M_ROWS   = 16 * 4096;         // B * S = 65536
constexpr int WAVES    = 8;                 // 256 threads, wave32
constexpr int M_TILE   = WAVES * 16;        // 128 rows per workgroup
constexpr int KSTEP    = 32;                // K per v_wmma_f32_16x16x32_f16
constexpr int NK       = IN_DIM / KSTEP;    // 8 K-steps
constexpr int NCOEFF   = 5;
constexpr int IC_FLOATS = IN_DIM * NCOEFF;  // 1280 floats

// LDS layout (dynamic shared): padded to dodge bank conflicts.
constexpr int XSTRIDE  = IN_DIM + 4;        // f32 stride: 260 dwords == 4 mod 64 banks
constexpr int WSTRIDE  = IN_DIM + 8;        // f16 stride: 264 halves (528 B, 16B-aligned rows)
constexpr int LDSX_BYTES = M_TILE * XSTRIDE * 4;            // 133120
constexpr int LDSW_BYTES = OUT_DIM * WSTRIDE * 2;           // 135168
constexpr int LDSC_BYTES = IC_FLOATS * 4;                   // 5120
constexpr int LDS_TOTAL  = LDSX_BYTES + LDSW_BYTES + LDSC_BYTES; // 273408 < 320KB
}

// tanh via native v_exp_f32 / v_rcp_f32: tanh(x) = 1 - 2/(e^{2x}+1).
__device__ __forceinline__ float fast_tanh(float v) {
    float e = __expf(2.0f * v);
    return 1.0f - 2.0f * __builtin_amdgcn_rcpf(e + 1.0f);
}

// Cubic Cox-de Boor on 8 uniform knots in [-1,1]; only basis 0 and 1 are active.
__device__ __forceinline__ void kan_basis(float x, float& b0, float& b1) {
    constexpr float h  = 2.0f / 7.0f;
    constexpr float t0 = -1.0f;
    constexpr float t1 = t0 + 1.0f * h;
    constexpr float t2 = t0 + 2.0f * h;
    constexpr float t3 = t0 + 3.0f * h;
    constexpr float t4 = t0 + 4.0f * h;
    constexpr float t5 = t0 + 5.0f * h;
    constexpr float ih  = 1.0f / h;
    constexpr float i2h = 0.5f * ih;
    constexpr float i3h = ih / 3.0f;

    const float N00 = (t0 <= x && x < t1) ? 1.0f : 0.0f;
    const float N01 = (t1 <= x && x < t2) ? 1.0f : 0.0f;
    const float N02 = (t2 <= x && x < t3) ? 1.0f : 0.0f;
    const float N03 = (t3 <= x && x < t4) ? 1.0f : 0.0f;
    const float N04 = (t4 <= x && x < t5) ? 1.0f : 0.0f;

    const float N10 = ((x - t0) * ih) * N00 + ((t2 - x) * ih) * N01;
    const float N11 = ((x - t1) * ih) * N01 + ((t3 - x) * ih) * N02;
    const float N12 = ((x - t2) * ih) * N02 + ((t4 - x) * ih) * N03;
    const float N13 = ((x - t3) * ih) * N03 + ((t5 - x) * ih) * N04;

    const float N20 = ((x - t0) * i2h) * N10 + ((t3 - x) * i2h) * N11;
    const float N21 = ((x - t1) * i2h) * N11 + ((t4 - x) * i2h) * N12;
    const float N22 = ((x - t2) * i2h) * N12 + ((t5 - x) * i2h) * N13;

    b0 = ((x - t0) * i3h) * N20 + ((t4 - x) * i3h) * N21;
    b1 = ((x - t1) * i3h) * N21 + ((t5 - x) * i3h) * N22;
}

__global__ __launch_bounds__(256)
void kan_fused_wmma(const float* __restrict__ x,
                    const float* __restrict__ ic,
                    const float* __restrict__ W,
                    float* __restrict__ out) {
    extern __shared__ __align__(16) char smem[];
    float*    lds_x  = (float*)smem;                                  // [128][260] f32
    _Float16* lds_w  = (_Float16*)(smem + LDSX_BYTES);                // [256][264] f16
    float*    lds_ic = (float*)(smem + LDSX_BYTES + LDSW_BYTES);      // [1280] f32

    const int tid  = threadIdx.x;
    const int lane = tid & 31;
    const int wave = tid >> 5;
    const int r    = lane & 15;     // row (A) / column (B,D) within 16x16 tile
    const int half = lane >> 4;     // K-chunk selector per WMMA operand layout

    const long long m_base = (long long)blockIdx.x * M_TILE;

    // ---- Stage x tile [128][256] f32 via Tensor Data Mover (one 2D D#) ----
    // TDM applies the +16B-per-1024B-row LDS padding itself (pad_interval=256
    // dwords -> code 7, pad_amount=4 dwords -> code 3), matching XSTRIDE=260.
    if (wave == 0) {
        const uint64_t gaddr    = (uint64_t)(uintptr_t)(x + m_base * IN_DIM);
        const unsigned lds_base = (unsigned)(uintptr_t)lds_x;

        u32x4 g0;
        g0[0] = 1u;                                               // count=1, user D#
        g0[1] = lds_base;                                         // lds_addr (bytes)
        g0[2] = (unsigned)(gaddr & 0xFFFFFFFFu);                  // global_addr[31:0]
        g0[3] = (unsigned)((gaddr >> 32) & 0x01FFFFFFu)           // global_addr[56:32]
                | (2u << 30);                                     // type = 2 ("image")

        u32x8 g1;
        g1[0] = (2u << 16)      // data_size = 4 bytes
              | (1u << 20)      // pad_enable
              | (7u << 22)      // pad_interval: 256 dwords (one 1KB row)
              | (3u << 25);     // pad_amount: 4 dwords (16 bytes)
        g1[1] = (unsigned)IN_DIM << 16;   // tensor_dim0[15:0] = 256
        g1[2] = (unsigned)M_TILE << 16;   // tensor_dim1[15:0] = 128 (tile-local, no OOB)
        g1[3] = (unsigned)IN_DIM << 16;   // tile_dim0 = 256
        g1[4] = (unsigned)M_TILE;         // tile_dim1 = 128, tile_dim2 = 0
        g1[5] = (unsigned)IN_DIM;         // tensor_dim0_stride = 256 elements
        g1[6] = 0u;                       // dim1_stride unused for 2D
        g1[7] = 0u;

        u32x4 gz = {0u, 0u, 0u, 0u};      // groups 2/3: unused (2D tensor)

        asm volatile("tensor_load_to_lds %0, %1, %2, %3"
                     :: "s"(g0), "s"(g1), "s"(gz), "s"(gz) : "memory");
    }

    // ---- Stage inner_coeffs into LDS: CDNA5 async copy (ASYNCcnt path) ----
    for (int c = tid; c < IC_FLOATS / 4; c += blockDim.x) {
        const float* gsrc = ic + c * 4;
        unsigned lds_off = (unsigned)(uintptr_t)(&lds_ic[c * 4]);
        asm volatile("global_load_async_to_lds_b128 %0, %1, off"
                     :: "v"(lds_off), "v"(gsrc) : "memory");
    }

    // ---- Stage W as f16 into LDS (convert once per workgroup, not per wave) ----
#pragma unroll
    for (int it = 0; it < (OUT_DIM * IN_DIM / 8) / 256; ++it) {     // 32 iters
        const int c    = it * 256 + tid;         // 8-element chunk id
        const int row  = c >> 5;                 // 32 chunks per row
        const int col8 = c & 31;
        f32x4 w0 = *(const f32x4*)(W + c * 8);
        f32x4 w1 = *(const f32x4*)(W + c * 8 + 4);
        v8h h;
#pragma unroll
        for (int e = 0; e < 4; ++e) { h[e] = (_Float16)w0[e]; h[4 + e] = (_Float16)w1[e]; }
        *(v8h*)(lds_w + row * WSTRIDE + col8 * 8) = h;
    }

    asm volatile("s_wait_asynccnt 0" ::: "memory");
    if (wave == 0) {
        __builtin_amdgcn_s_wait_tensorcnt(0);
    }
    __syncthreads();

    // ---- Build the 8 A-fragments (16x32 f16) in registers from LDS ----
    // A layout (ISA 7.12.2): lanes 0-15 -> M=lane, K {0..7,16..23};
    // lanes 16-31 -> M=lane-16, K {8..15,24..31}.
    const float* xr = lds_x + (wave * 16 + r) * XSTRIDE;

    v16h afrag[NK];
#pragma unroll
    for (int s = 0; s < NK; ++s) {
        const int i0 = s * KSTEP + half * 8;
        f32x4 xa0 = *(const f32x4*)(xr + i0);
        f32x4 xa1 = *(const f32x4*)(xr + i0 + 4);
        f32x4 xb0 = *(const f32x4*)(xr + i0 + 16);
        f32x4 xb1 = *(const f32x4*)(xr + i0 + 20);
#pragma unroll
        for (int e = 0; e < 8; ++e) {
            const int iA = i0 + e;
            const int iB = i0 + 16 + e;
            const float xa = (e < 4) ? xa0[e] : xa1[e - 4];
            const float xb = (e < 4) ? xb0[e] : xb1[e - 4];
            float b0, b1, innA, innB;
            kan_basis(fast_tanh(xa), b0, b1);
            innA = lds_ic[iA * NCOEFF + 0] * b0 + lds_ic[iA * NCOEFF + 1] * b1;
            kan_basis(fast_tanh(xb), b0, b1);
            innB = lds_ic[iB * NCOEFF + 0] * b0 + lds_ic[iB * NCOEFF + 1] * b1;
            afrag[s][e]     = (_Float16)innA;    // K = i0+e
            afrag[s][8 + e] = (_Float16)innB;    // K = i0+16+e
        }
    }

    // ---- GEMM: per 16-column tile, K-accumulate with WMMA; B from LDS f16 ----
    // B fragment (32x16 f16): lane l holds column n = nt*16 + l%16 = row n of W;
    // K chunks mirror the A layout -> two ds_load_b128 per fragment.
    for (int nt = 0; nt < OUT_DIM / 16; ++nt) {
        const int n = nt * 16 + r;
        const _Float16* wr = lds_w + n * WSTRIDE;

        v8f acc = {};
#pragma unroll
        for (int s = 0; s < NK; ++s) {
            const int i0 = s * KSTEP + half * 8;
            v8h lo = *(const v8h*)(wr + i0);        // K = i0..i0+7
            v8h hi = *(const v8h*)(wr + i0 + 16);   // K = i0+16..i0+23
            v16h b = __builtin_shufflevector(lo, hi,
                                             0, 1, 2, 3, 4, 5, 6, 7,
                                             8, 9, 10, 11, 12, 13, 14, 15);
            acc = __builtin_amdgcn_wmma_f32_16x16x32_f16(
                false, afrag[s], false, b, (short)0, acc, false, false);
        }

        // D layout (16x16 f32): lane l -> column l%16; VGPR v -> row v + 8*(l>=16).
        float* orow = out + (m_base + wave * 16 + half * 8) * OUT_DIM + nt * 16 + r;
#pragma unroll
        for (int v = 0; v < 8; ++v) {
            orow[(long long)v * OUT_DIM] = acc[v];
        }
    }
}

extern "C" void kernel_launch(void* const* d_in, const int* in_sizes, int n_in,
                              void* d_out, int out_size, void* d_ws, size_t ws_size,
                              hipStream_t stream) {
    (void)in_sizes; (void)n_in; (void)out_size; (void)d_ws; (void)ws_size;
    const float* x   = (const float*)d_in[0];   // [16, 4096, 256] f32
    const float* ic  = (const float*)d_in[1];   // [256, 5] f32
    const float* W   = (const float*)d_in[2];   // [256, 256] f32
    float*       out = (float*)d_out;           // [16, 4096, 256] f32

    dim3 grid(M_ROWS / M_TILE);   // 512 workgroups
    dim3 block(WAVES * 32);       // 256 threads = 8 wave32
    hipLaunchKernelGGL(kan_fused_wmma, grid, block, LDS_TOTAL, stream, x, ic, W, out);
}
